// Attention_88656714925438
// MI455X (gfx1250) — matrix operand order repaired and checked by reading, not compiled, not run on hardware
//
#include <hip/hip_runtime.h>
#include <hip/hip_bf16.h>

// ---------------------------------------------------------------------------
// DeBERTa disentangled attention, fused bf16 WMMA implementation for gfx1250.
// B=8, N=1024, C=384, H=6, D=64, SPAN=384. scale = sqrt(192), folded into Q
// and pos_q at projection time. Workspace usage: ~26.4 MB.
//
// All WMMA B-fragments are loaded with the contiguous-per-lane loader
// (ds_load_b128); tiles that need the transposed orientation (W in the GEMMs,
// V in the attention kernel) are stored transposed into LDS at tile-load time.
// ---------------------------------------------------------------------------

typedef __attribute__((ext_vector_type(16))) __bf16 bf16x16;
typedef __attribute__((ext_vector_type(8)))  float  f32x8;

#define WMMA_BF16(a, b, c) \
  __builtin_amdgcn_wmma_f32_16x16x32_bf16(false, (a), false, (b), (short)0, (c), false, false)

// A fragment, 16x32 bf16 (M x K), row-major source with leading dim ld.
// Layout: lane<16 -> row=lane, K=0..15;
//         lane>=16 -> row=lane-16, K=16..31 (matches load_bT).
static __device__ __forceinline__ bf16x16 load_a_frag(const __bf16* base, int ld) {
  const int lane = threadIdx.x & 31;
  const int row  = lane & 15;
  const int half = lane >> 4;
  const __bf16* p0 = base + row * ld + half * 16;
  const __bf16* p1 = p0 + 8;
  bf16x16 a;
#pragma unroll
  for (int t = 0; t < 8; ++t) a[t] = p0[t];
#pragma unroll
  for (int t = 0; t < 8; ++t) a[8 + t] = p1[t];
  return a;
}

// B fragment, 32x16 bf16 (K x N) built from a ROW-MAJOR [N][K] source
// (i.e. B = src^T). lane<16: N=lane, K=0..15; lane>=16: N=lane-16, K=16..31.
// Contiguous 32B read per lane -> ds_load_b128 pair.
static __device__ __forceinline__ bf16x16 load_bT(const __bf16* base, int ld) {
  const int lane = threadIdx.x & 31;
  const int col  = lane & 15;
  const int half = lane >> 4;
  const __bf16* p = base + col * ld + half * 16;
  bf16x16 b;
#pragma unroll
  for (int t = 0; t < 16; ++t) b[t] = p[t];
  return b;
}

// ---------------------------------------------------------------------------
// Generic 384-K GEMM: out[M,384] = (A[M,384] @ W[384,384] + bias) * alpha.
// 64x64 output tile per block, 128 threads (4 waves), bf16 WMMA, f32 accum.
// W tile is stored TRANSPOSED in LDS ([n][k]) so B-fragments are contiguous.
// ---------------------------------------------------------------------------
template <bool A_BF16, bool OUT_F32>
__global__ __launch_bounds__(128)
void gemm384_kernel(const void* __restrict__ Av, const float* __restrict__ W,
                    const float* __restrict__ bias, void* __restrict__ outv,
                    float alpha) {
  __shared__ __bf16 As[64 * 32];   // [m][k]
  __shared__ __bf16 WsT[64 * 32];  // [n][k]  (transposed W tile)
  const int m0 = blockIdx.x * 64, n0 = blockIdx.y * 64;
  const int tid = threadIdx.x, w = tid >> 5, lane = tid & 31;
  const int half = lane >> 4, ln = lane & 15;
  const float*  Af = (const float*)Av;
  const __bf16* Ab = (const __bf16*)Av;

  f32x8 acc[4];
#pragma unroll
  for (int t = 0; t < 4; ++t) acc[t] = (f32x8){0, 0, 0, 0, 0, 0, 0, 0};

  for (int k0 = 0; k0 < 384; k0 += 32) {
    __syncthreads();
    {
      const int row = tid & 63, c0 = (tid >> 6) * 16;
#pragma unroll
      for (int c = 0; c < 16; ++c) {
        float v = A_BF16 ? (float)Ab[(size_t)(m0 + row) * 384 + k0 + c0 + c]
                         : Af[(size_t)(m0 + row) * 384 + k0 + c0 + c];
        As[row * 32 + c0 + c] = (__bf16)v;
      }
      const int wr = tid & 31, wc0 = (tid >> 5) * 16;  // wr = k row, wc0 = n block
#pragma unroll
      for (int c = 0; c < 16; ++c)
        WsT[(wc0 + c) * 32 + wr] = (__bf16)W[(size_t)(k0 + wr) * 384 + n0 + wc0 + c];
    }
    __syncthreads();
    bf16x16 a = load_a_frag(As + 16 * w * 32, 32);
#pragma unroll
    for (int t = 0; t < 4; ++t) {
      bf16x16 b = load_bT(WsT + 16 * t * 32, 32);
      acc[t] = WMMA_BF16(a, b, acc[t]);
    }
  }
#pragma unroll
  for (int t = 0; t < 4; ++t) {
#pragma unroll
    for (int r = 0; r < 8; ++r) {
      const int row = m0 + 16 * w + r + 8 * half;
      const int col = n0 + 16 * t + ln;
      float v = (acc[t][r] + bias[col]) * alpha;
      if (OUT_F32) ((float*)outv)[(size_t)row * 384 + col] = v;
      else         ((__bf16*)outv)[(size_t)row * 384 + col] = (__bf16)v;
    }
  }
}

// ---------------------------------------------------------------------------
// Fused disentangled flash attention. One block = (b, h, 64-row i-tile).
// 128 threads (4 waves); wave w owns rows 16w..16w+15 of the i-tile.
// ---------------------------------------------------------------------------
__global__ __launch_bounds__(128)
void attn_kernel(const __bf16* __restrict__ Qg, const __bf16* __restrict__ Kg,
                 const __bf16* __restrict__ Vg, const __bf16* __restrict__ PKg,
                 const __bf16* __restrict__ PQg, __bf16* __restrict__ ctx) {
  extern __shared__ char smem[];
  float*  C2P = (float*)smem;               // 64 x 128
  float*  P2C = C2P + 64 * 128;             // 64 x 128
  __bf16* Qs  = (__bf16*)(P2C + 64 * 128);  // 64 x 64  [i][d]
  __bf16* Ks  = Qs + 64 * 64;               // 64 x 64  [j][d]
  __bf16* VsT = Ks + 64 * 64;               // 64 x 64  [d][j]  (transposed V)
  __bf16* Pk  = VsT + 64 * 64;              // 128 x 64 pos_k slice [t][d]
  __bf16* Pq  = Pk + 128 * 64;              // 128 x 64 pos_q slice [t][d]
  __bf16* Ps  = Pq + 128 * 64;              // 64 x 64 softmax probs (bf16)

  const int it = blockIdx.x & 15;
  const int h  = (blockIdx.x >> 4) % 6;
  const int b  = blockIdx.x / 96;
  const int i0 = it * 64;
  const int tid = threadIdx.x, w = tid >> 5, lane = tid & 31;
  const int half = lane >> 4, ln = lane & 15;

  const f32x8 vzero = {0, 0, 0, 0, 0, 0, 0, 0};

  // Load Q tile (64 x 64) into LDS.
  {
    const int row = tid & 63, c0 = (tid >> 6) * 32;
    const size_t gb = ((size_t)b * 1024 + i0 + row) * 384 + h * 64 + c0;
#pragma unroll
    for (int c = 0; c < 32; ++c) Qs[row * 64 + c0 + c] = Qg[gb + c];
  }
  __syncthreads();
  const bf16x16 aq0 = load_a_frag(Qs + 16 * w * 64, 64);       // K = d 0..31
  const bf16x16 aq1 = load_a_frag(Qs + 16 * w * 64 + 32, 64);  // K = d 32..63

  f32x8 O[4];
  float mrun[8], lrun[8];
#pragma unroll
  for (int t = 0; t < 4; ++t) O[t] = vzero;
#pragma unroll
  for (int r = 0; r < 8; ++r) { mrun[r] = -1e30f; lrun[r] = 0.f; }

  for (int jt = 0; jt < 16; ++jt) {
    const int j0 = jt * 64;
    const int sbase = min(max(i0 - j0 + 321, 0), 640);  // slice start index
    __syncthreads();  // previous iteration's LDS consumers done
    {
      const int row = tid & 63, c0 = (tid >> 6) * 32;
      const size_t gb = ((size_t)b * 1024 + j0 + row) * 384 + h * 64 + c0;
#pragma unroll
      for (int c = 0; c < 32; ++c) Ks[row * 64 + c0 + c] = Kg[gb + c];
      // V stored transposed: VsT[d][j]
#pragma unroll
      for (int c = 0; c < 32; ++c) VsT[(c0 + c) * 64 + row] = Vg[gb + c];
      // pos slices: one 128B row per thread (128 rows each)
      const size_t pb = ((size_t)(sbase + tid)) * 384 + h * 64;
#pragma unroll
      for (int c = 0; c < 64; ++c) Pk[tid * 64 + c] = PKg[pb + c];
#pragma unroll
      for (int c = 0; c < 64; ++c) Pq[tid * 64 + c] = PQg[pb + c];
      if (jt < 15)
        __builtin_prefetch(&Kg[((size_t)b * 1024 + j0 + 64 + row) * 384 + h * 64 + c0], 0, 0);
    }
    __syncthreads();

    // S = (q/s) k^T : 16 x 64 per wave
    f32x8 S[4];
#pragma unroll
    for (int t = 0; t < 4; ++t) {
      f32x8 s = vzero;
      s = WMMA_BF16(aq0, load_bT(Ks + 16 * t * 64, 64), s);
      s = WMMA_BF16(aq1, load_bT(Ks + 16 * t * 64 + 32, 64), s);
      S[t] = s;
    }
    // c2p slice: rows(i) 16w..16w+15  x  128 pos columns -> LDS
#pragma unroll
    for (int t = 0; t < 8; ++t) {
      f32x8 c = vzero;
      c = WMMA_BF16(aq0, load_bT(Pk + 16 * t * 64, 64), c);
      c = WMMA_BF16(aq1, load_bT(Pk + 16 * t * 64 + 32, 64), c);
#pragma unroll
      for (int r = 0; r < 8; ++r)
        C2P[(16 * w + r + 8 * half) * 128 + 16 * t + ln] = c[r];
    }
    // p2c slice: rows(j) 16w..16w+15  x  128 pos columns -> LDS
    const bf16x16 ak0 = load_a_frag(Ks + 16 * w * 64, 64);
    const bf16x16 ak1 = load_a_frag(Ks + 16 * w * 64 + 32, 64);
#pragma unroll
    for (int t = 0; t < 8; ++t) {
      f32x8 c = vzero;
      c = WMMA_BF16(ak0, load_bT(Pq + 16 * t * 64, 64), c);
      c = WMMA_BF16(ak1, load_bT(Pq + 16 * t * 64 + 32, 64), c);
#pragma unroll
      for (int r = 0; r < 8; ++r)
        P2C[(16 * w + r + 8 * half) * 128 + 16 * t + ln] = c[r];
    }
    __syncthreads();

    // Diagonal gather of relative-position bias into S.
#pragma unroll
    for (int t = 0; t < 4; ++t) {
#pragma unroll
      for (int r = 0; r < 8; ++r) {
        const int il = 16 * w + r + 8 * half;
        const int jl = 16 * t + ln;
        const int tt = min(max((i0 + il) - (j0 + jl) + 384, 0), 767);
        const int col = tt - sbase;  // guaranteed in [0,127]
        S[t][r] += C2P[il * 128 + col] + P2C[jl * 128 + col];
      }
    }

    // Online softmax (per-row stats; 16-lane half-wave reductions).
#pragma unroll
    for (int r = 0; r < 8; ++r) {
      float v = fmaxf(fmaxf(S[0][r], S[1][r]), fmaxf(S[2][r], S[3][r]));
#pragma unroll
      for (int m = 1; m <= 8; m <<= 1) v = fmaxf(v, __shfl_xor(v, m, 32));
      const float mnew = fmaxf(mrun[r], v);
      const float sf = __expf(mrun[r] - mnew);
      mrun[r] = mnew;
#pragma unroll
      for (int t = 0; t < 4; ++t) O[t][r] *= sf;
      float s = 0.f;
#pragma unroll
      for (int t = 0; t < 4; ++t) { S[t][r] = __expf(S[t][r] - mnew); s += S[t][r]; }
#pragma unroll
      for (int m = 1; m <= 8; m <<= 1) s += __shfl_xor(s, m, 32);
      lrun[r] = lrun[r] * sf + s;
    }

    // P (bf16) via wave-private LDS round-trip, then O += P @ V.
    __bf16* Pw = Ps + 16 * w * 64;
#pragma unroll
    for (int t = 0; t < 4; ++t)
#pragma unroll
      for (int r = 0; r < 8; ++r)
        Pw[(r + 8 * half) * 64 + 16 * t + ln] = (__bf16)S[t][r];
    asm volatile("s_wait_dscnt 0" ::: "memory");
    const bf16x16 pa0 = load_a_frag(Pw, 64);       // K = j 0..31
    const bf16x16 pa1 = load_a_frag(Pw + 32, 64);  // K = j 32..63
#pragma unroll
    for (int t = 0; t < 4; ++t) {
      O[t] = WMMA_BF16(pa0, load_bT(VsT + 16 * t * 64, 64), O[t]);
      O[t] = WMMA_BF16(pa1, load_bT(VsT + 16 * t * 64 + 32, 64), O[t]);
    }
  }

  // Normalize and write context (bf16, [b, n, h*64+d]).
#pragma unroll
  for (int t = 0; t < 4; ++t)
#pragma unroll
    for (int r = 0; r < 8; ++r) {
      const int il = 16 * w + r + 8 * half;
      const int d  = 16 * t + ln;
      ctx[((size_t)b * 1024 + i0 + il) * 384 + h * 64 + d] =
          (__bf16)(O[t][r] * (1.0f / lrun[r]));
    }
}

// ---------------------------------------------------------------------------
extern "C" void kernel_launch(void* const* d_in, const int* in_sizes, int n_in,
                              void* d_out, int out_size, void* d_ws, size_t ws_size,
                              hipStream_t stream) {
  (void)in_sizes; (void)n_in; (void)out_size; (void)ws_size;
  const float* x   = (const float*)d_in[0];
  // d_in[1] relative_pos: implied by (i - j); not needed.
  const float* rel = (const float*)d_in[2];
  const float* Wq  = (const float*)d_in[3];  const float* bq  = (const float*)d_in[4];
  const float* Wk  = (const float*)d_in[5];  const float* bk  = (const float*)d_in[6];
  const float* Wv  = (const float*)d_in[7];  const float* bv  = (const float*)d_in[8];
  const float* Wpk = (const float*)d_in[9];  const float* bpk = (const float*)d_in[10];
  const float* Wpq = (const float*)d_in[11]; const float* bpq = (const float*)d_in[12];
  const float* Wo  = (const float*)d_in[13]; const float* bo  = (const float*)d_in[14];

  // Workspace carve-up (bf16): Q,K,V,ctx (8192x384 each) + pos_k,pos_q (768x384).
  __bf16* Qb = (__bf16*)d_ws;
  __bf16* Kb = Qb + (size_t)8192 * 384;
  __bf16* Vb = Kb + (size_t)8192 * 384;
  __bf16* Cx = Vb + (size_t)8192 * 384;
  __bf16* Pk = Cx + (size_t)8192 * 384;
  __bf16* Pq = Pk + (size_t)768 * 384;

  const float inv_scale = 1.0f / sqrtf(192.0f);  // 1/sqrt(D * SCALE_FACTOR)
  dim3 blk(128);
  dim3 gx(128, 6);   // M=8192
  dim3 gp(12, 6);    // M=768

  gemm384_kernel<false, false><<<gx, blk, 0, stream>>>(x,   Wq,  bq,  Qb, inv_scale);
  gemm384_kernel<false, false><<<gx, blk, 0, stream>>>(x,   Wk,  bk,  Kb, 1.0f);
  gemm384_kernel<false, false><<<gx, blk, 0, stream>>>(x,   Wv,  bv,  Vb, 1.0f);
  gemm384_kernel<false, false><<<gp, blk, 0, stream>>>(rel, Wpk, bpk, Pk, 1.0f);
  gemm384_kernel<false, false><<<gp, blk, 0, stream>>>(rel, Wpq, bpq, Pq, inv_scale);

  const size_t smem = (size_t)(64 * 128 * 4) * 2   // C2P, P2C (f32)
                    + (size_t)(64 * 64 * 2) * 4    // Qs, Ks, VsT, Ps (bf16)
                    + (size_t)(128 * 64 * 2) * 2;  // Pk, Pq slices (bf16)
  attn_kernel<<<dim3(8 * 6 * 16), blk, smem, stream>>>(Qb, Kb, Vb, Pk, Pq, Cx);

  gemm384_kernel<true, true><<<gx, blk, 0, stream>>>(Cx, Wo, bo, (float*)d_out, 1.0f);
}
